// LocalWLNet_12816182411668
// MI455X (gfx1250) — compile-verified, matrix operand order checked by reading, not run on hardware
//
#include <hip/hip_runtime.h>

#define HID1 256
#define HID2 32

typedef __attribute__((ext_vector_type(16))) __bf16 v16bf;
typedef __attribute__((ext_vector_type(8)))  float  v8f;

union Frag16 { v16bf v; __bf16 e[16]; };
union BFrag  { uint4 u[2]; v16bf v; };          // 32 B: one lane's B fragment

__device__ __forceinline__ unsigned short bf16u(float f) {
    // round-to-nearest-even f32 -> bf16 bits
    unsigned u = __builtin_bit_cast(unsigned, f);
    unsigned r = u + 0x7FFFu + ((u >> 16) & 1u);
    return (unsigned short)(r >> 16);
}
__device__ __forceinline__ __bf16 f2bf(float f) {
    unsigned short h = bf16u(f);
    return __builtin_bit_cast(__bf16, h);
}
__device__ __forceinline__ __bf16 u2bf(unsigned short h) {
    return __builtin_bit_cast(__bf16, h);
}

// ---------------- weight pre-pack: W[K x 32] f32 -> bf16 WMMA fragments -------
// layout: frag[((kk*2 + c)*32 + lane)*16 + j], K = kk*32 + (j>>3)*16 + (lane>>4)*8 + (j&7),
//         n = (lane&15) + 16*c.  One lane's fragment = 32 contiguous bytes.

__global__ void k_pack_wfrag(const float* __restrict__ W, unsigned short* __restrict__ out,
                             int ktot) {
    int t = blockIdx.x * blockDim.x + threadIdx.x;
    if (t >= ktot * HID2) return;
    int j    = t & 15;
    int lane = (t >> 4) & 31;
    int c    = (t >> 9) & 1;
    int kk   = t >> 10;
    int K = kk * 32 + ((j >> 3) << 4) + ((lane >> 4) << 3) + (j & 7);
    int n = (lane & 15) + (c << 4);
    out[t] = bf16u(W[K * HID2 + n]);
}

// ---------------- degree / normalization ----------------

__global__ void k_deg_init(float* __restrict__ d1, int n,
                           float* __restrict__ d2a, float* __restrict__ d2b, int m) {
    int i = blockIdx.x * blockDim.x + threadIdx.x;
    if (i < n) d1[i] = 1.0f;                 // self-loop
    if (i < m) { d2a[i] = 1.0f; d2b[i] = 1.0f; }
}

__global__ void k_deg_count1(const int* __restrict__ dst, float* __restrict__ deg, int e) {
    int i = blockIdx.x * blockDim.x + threadIdx.x;
    if (i < e) atomicAdd(&deg[dst[i]], 1.0f);
}

__global__ void k_deg_count2(const int* __restrict__ ei2,
                             float* __restrict__ dega, float* __restrict__ degb, int e) {
    int i = blockIdx.x * blockDim.x + threadIdx.x;
    if (i < e) {
        atomicAdd(&dega[ei2[(size_t)e + i]], 1.0f);  // conv2a dst = ei2[1]
        atomicAdd(&degb[ei2[i]], 1.0f);              // conv2b dst = ei2[0]
    }
}

__global__ void k_rsqrt(float* __restrict__ d1, int n,
                        float* __restrict__ d2a, float* __restrict__ d2b, int m) {
    int i = blockIdx.x * blockDim.x + threadIdx.x;
    if (i < n) d1[i] = rsqrtf(d1[i]);
    if (i < m) { d2a[i] = rsqrtf(d2a[i]); d2b[i] = rsqrtf(d2b[i]); }
}

// ---------------- GEMM 1: xw1 = emb_table[x] @ W1  (WMMA bf16) ----------------
// wave = one 16-row block; A via 4x b128 loads/step; B from pre-packed fragments.

__global__ void k_emb_gemm1(const int* __restrict__ x, const float* __restrict__ emb,
                            const unsigned short* __restrict__ w1f,
                            float* __restrict__ xw1, int nblk, int n) {
    int wave = (int)((blockIdx.x * (unsigned)blockDim.x + threadIdx.x) >> 5);
    if (wave >= nblk) return;                      // wave-uniform: EXEC stays full
    int lane = threadIdx.x & 31;
    int m    = lane & 15;
    int half = lane >> 4;
    int row  = wave * 16 + m;
    if (row >= n) row = n - 1;                     // clamp (N divisible by 16 anyway)
    const float* arow = emb + (size_t)x[row] * HID1;

    v8f c0 = {}; v8f c1 = {};
    #pragma unroll
    for (int kk = 0; kk < HID1 / 32; ++kk) {
        int kb = kk * 32 + (half << 3);
        float4 a0 = *(const float4*)(arow + kb);
        float4 a1 = *(const float4*)(arow + kb + 4);
        float4 a2 = *(const float4*)(arow + kb + 16);
        float4 a3 = *(const float4*)(arow + kb + 20);
        Frag16 a;
        a.e[0]  = f2bf(a0.x); a.e[1]  = f2bf(a0.y); a.e[2]  = f2bf(a0.z); a.e[3]  = f2bf(a0.w);
        a.e[4]  = f2bf(a1.x); a.e[5]  = f2bf(a1.y); a.e[6]  = f2bf(a1.z); a.e[7]  = f2bf(a1.w);
        a.e[8]  = f2bf(a2.x); a.e[9]  = f2bf(a2.y); a.e[10] = f2bf(a2.z); a.e[11] = f2bf(a2.w);
        a.e[12] = f2bf(a3.x); a.e[13] = f2bf(a3.y); a.e[14] = f2bf(a3.z); a.e[15] = f2bf(a3.w);

        const uint4* p0 = (const uint4*)(w1f + (size_t)(((kk * 2 + 0) * 32 + lane) << 4));
        const uint4* p1 = (const uint4*)(w1f + (size_t)(((kk * 2 + 1) * 32 + lane) << 4));
        BFrag b0, b1;
        b0.u[0] = p0[0]; b0.u[1] = p0[1];
        b1.u[0] = p1[0]; b1.u[1] = p1[1];

        c0 = __builtin_amdgcn_wmma_f32_16x16x32_bf16(false, a.v, false, b0.v,
                                                     (short)0, c0, false, false);
        c1 = __builtin_amdgcn_wmma_f32_16x16x32_bf16(false, a.v, false, b1.v,
                                                     (short)0, c1, false, false);
    }
    #pragma unroll
    for (int r = 0; r < 8; ++r) {
        int orow = wave * 16 + (half << 3) + r;
        if (orow < n) {
            xw1[(size_t)orow * HID2 + m]      = c0[r];
            xw1[(size_t)orow * HID2 + m + 16] = c1[r];
        }
    }
}

// ---------------- GEMM 2: xw2{a,b} = hp(bf16) @ W2{a,b} (K=32, one step) ------

__global__ void k_pair_gemm2(const unsigned short* __restrict__ hp,
                             const unsigned short* __restrict__ w2af,
                             const unsigned short* __restrict__ w2bf,
                             float* __restrict__ xw2a, float* __restrict__ xw2b,
                             int nblk, int np) {
    int wave = (int)((blockIdx.x * (unsigned)blockDim.x + threadIdx.x) >> 5);
    if (wave >= nblk) return;
    int lane = threadIdx.x & 31;
    int m    = lane & 15;
    int half = lane >> 4;
    int row  = wave * 16 + m;
    if (row >= np) row = np - 1;
    const unsigned short* arow = hp + (size_t)row * HID2;

    union { uint4 u; unsigned short s[8]; } r0, r1;
    r0.u = *(const uint4*)(arow + (half << 3));
    r1.u = *(const uint4*)(arow + 16 + (half << 3));
    Frag16 a;
    #pragma unroll
    for (int j = 0; j < 8; ++j) {
        a.e[j]     = u2bf(r0.s[j]);
        a.e[j + 8] = u2bf(r1.s[j]);
    }

    BFrag ba0, ba1, bb0, bb1;
    {
        const uint4* pa0 = (const uint4*)(w2af + (size_t)(((0 * 32 + lane)) << 4));
        const uint4* pa1 = (const uint4*)(w2af + (size_t)(((1 * 32 + lane)) << 4));
        const uint4* pb0 = (const uint4*)(w2bf + (size_t)(((0 * 32 + lane)) << 4));
        const uint4* pb1 = (const uint4*)(w2bf + (size_t)(((1 * 32 + lane)) << 4));
        ba0.u[0] = pa0[0]; ba0.u[1] = pa0[1];
        ba1.u[0] = pa1[0]; ba1.u[1] = pa1[1];
        bb0.u[0] = pb0[0]; bb0.u[1] = pb0[1];
        bb1.u[0] = pb1[0]; bb1.u[1] = pb1[1];
    }
    v8f z = {};
    v8f ca0 = __builtin_amdgcn_wmma_f32_16x16x32_bf16(false, a.v, false, ba0.v, (short)0, z, false, false);
    v8f ca1 = __builtin_amdgcn_wmma_f32_16x16x32_bf16(false, a.v, false, ba1.v, (short)0, z, false, false);
    v8f cb0 = __builtin_amdgcn_wmma_f32_16x16x32_bf16(false, a.v, false, bb0.v, (short)0, z, false, false);
    v8f cb1 = __builtin_amdgcn_wmma_f32_16x16x32_bf16(false, a.v, false, bb1.v, (short)0, z, false, false);

    #pragma unroll
    for (int r = 0; r < 8; ++r) {
        int orow = wave * 16 + (half << 3) + r;
        if (orow < np) {
            xw2a[(size_t)orow * HID2 + m]      = ca0[r];
            xw2a[(size_t)orow * HID2 + m + 16] = ca1[r];
            xw2b[(size_t)orow * HID2 + m]      = cb0[r];
            xw2b[(size_t)orow * HID2 + m + 16] = cb1[r];
        }
    }
}

// ---------------- GCN self-loop init / scatter / epilogue ----------------

__global__ void k_self1(const float* __restrict__ xw, const float* __restrict__ dinv,
                        float* __restrict__ acc, int n) {
    int t = blockIdx.x * blockDim.x + threadIdx.x;
    if (t >= n * HID2) return;
    float di = dinv[t >> 5];
    acc[t] = xw[t] * di * di;
}

// one wave per edge, lane = feature: coalesced gather + coalesced float atomics
__global__ void k_scatter1(const int* __restrict__ edge, const float* __restrict__ dinv,
                           const float* __restrict__ xw, float* __restrict__ acc, int e) {
    long t  = (long)blockIdx.x * blockDim.x + threadIdx.x;
    long ed = t >> 5;
    if (ed >= e) return;
    int k = (int)(t & 31);
    int s = edge[ed];
    int d = edge[(size_t)e + ed];
    float w = dinv[s] * dinv[d];
    atomicAdd(&acc[(size_t)d * HID2 + k], xw[(size_t)s * HID2 + k] * w);
}

__global__ void k_bias_relu(float* __restrict__ h, const float* __restrict__ b, int n) {
    int t = blockIdx.x * blockDim.x + threadIdx.x;
    if (t >= n * HID2) return;
    h[t] = fmaxf(h[t] + b[t & 31], 0.0f);
}

// pair gather-multiply, store bf16 (hp feeds the WMMA A fragments directly)
__global__ void k_pair_mul(const float* __restrict__ h1, const int* __restrict__ pos,
                           unsigned short* __restrict__ hp, int np) {
    int t = blockIdx.x * blockDim.x + threadIdx.x;
    if (t >= np * HID2) return;
    int m = t >> 5, k = t & 31;
    int a = pos[2 * m], b = pos[2 * m + 1];
    hp[t] = bf16u(h1[(size_t)a * HID2 + k] * h1[(size_t)b * HID2 + k]);
}

__global__ void k_self2(const float* __restrict__ xwa, const float* __restrict__ dva,
                        float* __restrict__ acca,
                        const float* __restrict__ xwb, const float* __restrict__ dvb,
                        float* __restrict__ accb, int np) {
    int t = blockIdx.x * blockDim.x + threadIdx.x;
    if (t >= np * HID2) return;
    int i = t >> 5;
    float da = dva[i]; acca[t] = xwa[t] * da * da;
    float db = dvb[i]; accb[t] = xwb[t] * db * db;
}

// fused forward + reverse scatter: reads edge list once
__global__ void k_scatter2(const int* __restrict__ ei2,
                           const float* __restrict__ dva, const float* __restrict__ dvb,
                           const float* __restrict__ xwa, const float* __restrict__ xwb,
                           float* __restrict__ acca, float* __restrict__ accb, int e) {
    long t  = (long)blockIdx.x * blockDim.x + threadIdx.x;
    long ed = t >> 5;
    if (ed >= e) return;
    int k = (int)(t & 31);
    int u = ei2[ed];                 // ei2[0][e]
    int v = ei2[(size_t)e + ed];     // ei2[1][e]
    float wa = dva[u] * dva[v];      // conv2a: s=u -> d=v
    atomicAdd(&acca[(size_t)v * HID2 + k], xwa[(size_t)u * HID2 + k] * wa);
    float wb = dvb[v] * dvb[u];      // conv2b: s=v -> d=u
    atomicAdd(&accb[(size_t)u * HID2 + k], xwb[(size_t)v * HID2 + k] * wb);
}

// fused: relu(acc_a+b2a)+relu(acc_b+b2b), gather idx pairs, multiply, dot Wp, +bp
__global__ void k_final(const float* __restrict__ acca, const float* __restrict__ accb,
                        const float* __restrict__ b2a, const float* __restrict__ b2b,
                        const int* __restrict__ idx, const float* __restrict__ Wp,
                        const float* __restrict__ bp, float* __restrict__ out, int q) {
    long t = (long)blockIdx.x * blockDim.x + threadIdx.x;
    long qq = t >> 5;
    if (qq >= q) return;
    int k = (int)(t & 31);
    int i0 = idx[2 * qq];
    int i1 = idx[2 * qq + 1];
    float ba = b2a[k], bb = b2b[k];
    float v0 = fmaxf(acca[(size_t)i0 * HID2 + k] + ba, 0.0f)
             + fmaxf(accb[(size_t)i0 * HID2 + k] + bb, 0.0f);
    float v1 = fmaxf(acca[(size_t)i1 * HID2 + k] + ba, 0.0f)
             + fmaxf(accb[(size_t)i1 * HID2 + k] + bb, 0.0f);
    float p = v0 * v1 * Wp[k];
    #pragma unroll
    for (int off = 16; off > 0; off >>= 1) p += __shfl_xor(p, off, 32);
    if (k == 0) out[qq] = p + bp[0];
}

// ---------------- launch ----------------

static inline int cdiv(long a, long b) { return (int)((a + b - 1) / b); }

extern "C" void kernel_launch(void* const* d_in, const int* in_sizes, int n_in,
                              void* d_out, int out_size, void* d_ws, size_t ws_size,
                              hipStream_t stream) {
    (void)n_in; (void)ws_size;
    const int*   x     = (const int*)d_in[0];
    const int*   edge1 = (const int*)d_in[1];
    const int*   pos   = (const int*)d_in[2];
    const int*   idx   = (const int*)d_in[3];
    const int*   ei2   = (const int*)d_in[4];
    const float* emb   = (const float*)d_in[5];
    const float* W1    = (const float*)d_in[6];
    const float* b1    = (const float*)d_in[7];
    const float* W2a   = (const float*)d_in[8];
    const float* b2a   = (const float*)d_in[9];
    const float* W2b   = (const float*)d_in[10];
    const float* b2b   = (const float*)d_in[11];
    const float* Wp    = (const float*)d_in[12];
    const float* bp    = (const float*)d_in[13];
    float*       out   = (float*)d_out;

    const int N  = in_sizes[0];        // 50000 nodes
    const int E1 = in_sizes[1] / 2;    // 1.6M edges
    const int NP = in_sizes[3];        // 200000 pairs
    const int E2 = in_sizes[4] / 2;    // 1.6M pair-graph edges
    const int Q  = out_size;           // 100000 outputs

    // carve workspace with 256 B alignment (vector-load safe)
    char* p = (char*)d_ws;
    auto carve = [&](size_t bytes) -> void* {
        void* r = (void*)p;
        p += (bytes + 255) & ~(size_t)255;
        return r;
    };
    float*          xw1    = (float*)carve((size_t)N  * HID2 * 4);
    float*          h1     = (float*)carve((size_t)N  * HID2 * 4);
    float*          dinv1  = (float*)carve((size_t)N * 4);
    unsigned short* hp     = (unsigned short*)carve((size_t)NP * HID2 * 2);
    float*          xw2a   = (float*)carve((size_t)NP * HID2 * 4);
    float*          xw2b   = (float*)carve((size_t)NP * HID2 * 4);
    float*          acca   = (float*)carve((size_t)NP * HID2 * 4);
    float*          accb   = (float*)carve((size_t)NP * HID2 * 4);
    float*          dinv2a = (float*)carve((size_t)NP * 4);
    float*          dinv2b = (float*)carve((size_t)NP * 4);
    unsigned short* w1f    = (unsigned short*)carve((size_t)HID1 * HID2 * 2);
    unsigned short* w2af   = (unsigned short*)carve((size_t)HID2 * HID2 * 2);
    unsigned short* w2bf   = (unsigned short*)carve((size_t)HID2 * HID2 * 2);

    const int B = 256;
    int mx = (N > NP) ? N : NP;

    // weight pre-pack (tiny, one-shot per call)
    k_pack_wfrag<<<cdiv(HID1 * HID2, B), B, 0, stream>>>(W1,  w1f,  HID1);
    k_pack_wfrag<<<cdiv(HID2 * HID2, B), B, 0, stream>>>(W2a, w2af, HID2);
    k_pack_wfrag<<<cdiv(HID2 * HID2, B), B, 0, stream>>>(W2b, w2bf, HID2);

    k_deg_init  <<<cdiv(mx, B), B, 0, stream>>>(dinv1, N, dinv2a, dinv2b, NP);
    k_deg_count1<<<cdiv(E1, B), B, 0, stream>>>(edge1 + E1, dinv1, E1);
    k_deg_count2<<<cdiv(E2, B), B, 0, stream>>>(ei2, dinv2a, dinv2b, E2);
    k_rsqrt     <<<cdiv(mx, B), B, 0, stream>>>(dinv1, N, dinv2a, dinv2b, NP);

    int nblk1 = (N + 15) / 16;
    k_emb_gemm1<<<cdiv((long)nblk1 * 32, B), B, 0, stream>>>(x, emb, w1f, xw1, nblk1, N);

    k_self1    <<<cdiv((long)N * HID2, B), B, 0, stream>>>(xw1, dinv1, h1, N);
    k_scatter1 <<<cdiv((long)E1 * 32, B), B, 0, stream>>>(edge1, dinv1, xw1, h1, E1);
    k_bias_relu<<<cdiv((long)N * HID2, B), B, 0, stream>>>(h1, b1, N);

    k_pair_mul <<<cdiv((long)NP * HID2, B), B, 0, stream>>>(h1, pos, hp, NP);

    int nblk2 = (NP + 15) / 16;
    k_pair_gemm2<<<cdiv((long)nblk2 * 32, B), B, 0, stream>>>(hp, w2af, w2bf, xw2a, xw2b, nblk2, NP);

    k_self2    <<<cdiv((long)NP * HID2, B), B, 0, stream>>>(xw2a, dinv2a, acca, xw2b, dinv2b, accb, NP);
    k_scatter2 <<<cdiv((long)E2 * 32, B), B, 0, stream>>>(ei2, dinv2a, dinv2b, xw2a, xw2b, acca, accb, E2);

    k_final    <<<cdiv((long)Q * 32, B), B, 0, stream>>>(acca, accb, b2a, b2b, idx, Wp, bp, out, Q);
}
